// Decoder_featurizer_70626442215582
// MI455X (gfx1250) — matrix operand order repaired and checked by reading, not compiled, not run on hardware
//
#include <hip/hip_runtime.h>
#include <hip/hip_bf16.h>

// Problem geometry
#define HRR    1024
#define NDEC   16
#define BATCH  8192
#define OUTW   (2 * HRR + 2 * NDEC * HRR)          // 34816 f32 per batch row
#define SIDE_ELEMS ((size_t)BATCH * HRR)           // 8,388,608 per side
#define W_ELEMS    ((size_t)NDEC * HRR * HRR)      // 16,777,216  (Wt: [16384][1024])

// CDNA5 WMMA operand types (wave32): 16x16x32 bf16 -> f32
typedef __attribute__((ext_vector_type(16))) __bf16 v16bf;
typedef __attribute__((ext_vector_type(8)))  float  v8f;

// Trivial POD vectors (HIP's float4/uint4 have ctors that break unions)
struct alignas(16) U4 { unsigned int x, y, z, w; };
struct alignas(8)  U2 { unsigned int x, y; };
struct alignas(16) F4 { float x, y, z, w; };

// ---------------------------------------------------------------------------
// Kernel 1: convert X (both sides) f32 -> bf16 into ws, and write the f32
// passthrough columns [0,2048) of the output in the same pass.
// 4 elements / thread, 4,194,304 threads.
// ---------------------------------------------------------------------------
__global__ __launch_bounds__(256) void xcvt_kernel(const float* __restrict__ prob,
                                                   const float* __restrict__ lem,
                                                   __bf16* __restrict__ xbf,
                                                   float* __restrict__ out) {
  const size_t t = (size_t)blockIdx.x * 256 + threadIdx.x;
  const size_t e = t * 4;                       // flat index into [2][BATCH][HRR]
  const int    side = (e >= SIDE_ELEMS) ? 1 : 0;
  const size_t r = e - (side ? SIDE_ELEMS : 0); // b*HRR + m
  const size_t b = r >> 10;
  const int    m = (int)(r & (HRR - 1));

  const F4 f = *(const F4*)((side ? lem : prob) + r);
  *(F4*)(out + b * OUTW + (size_t)side * HRR + m) = f;   // concat passthrough

  union { __bf16 h[4]; U2 u; } p;
  p.h[0] = (__bf16)f.x; p.h[1] = (__bf16)f.y;
  p.h[2] = (__bf16)f.z; p.h[3] = (__bf16)f.w;
  *(U2*)(xbf + e) = p.u;
}

// ---------------------------------------------------------------------------
// Kernel 2: materialize Wt[j][m] = bf16(dec[d][(m-k) & 1023]), j = d*1024 + k.
// K-contiguous (transposed) so GEMM B-fragments are contiguous b128 loads.
// Reads the 64KB decoder table (cache-hot), writes 32 MB bf16.
// ---------------------------------------------------------------------------
__global__ __launch_bounds__(256) void wgen_kernel(const float* __restrict__ dec,
                                                   __bf16* __restrict__ wt) {
  const size_t t = (size_t)blockIdx.x * 256 + threadIdx.x;
  const size_t e = t * 4;                 // flat index into [16384][1024]
  const int j  = (int)(e >> 10);
  const int m0 = (int)(e & (HRR - 1));
  const int d  = j >> 10;
  const int k  = j & (HRR - 1);
  const float* dr = dec + (size_t)d * HRR;

  union { __bf16 h[4]; U2 u; } p;
#pragma unroll
  for (int i = 0; i < 4; ++i)
    p.h[i] = (__bf16)dr[(m0 + i - k) & (HRR - 1)];
  *(U2*)(wt + e) = p.u;
}

// ---------------------------------------------------------------------------
// Kernel 3: GEMM  C[b, j] = sum_m Xbf[side][b][m] * Wt[j][m]
// Block = 256 thr = 8 waves in a 2(M) x 4(N) grid; block tile 64x128.
// Wave tile 32x32 = 2x2 v_wmma_f32_16x16x32_bf16 accumulators; K = 1024.
// Fragment layouts per ISA 7.12.2 (wave32):
//   A 16x32 bf16: lanes 0-15 -> M=lane, K {0..7, 16..23}; lanes 16-31 -> K {8..15, 24..31}
//   B 32x16 bf16: lanes 0-15 -> N=lane, K {0..7, 16..23}; lanes 16-31 -> K {8..15, 24..31}
//   C f32: vgpr v -> M = v + 8*(lane>=16), N = lane&15
// ---------------------------------------------------------------------------
__global__ __launch_bounds__(256) void hrr_gemm_kernel(const __bf16* __restrict__ xbf,
                                                       const __bf16* __restrict__ wt,
                                                       float* __restrict__ out) {
  const int lane  = threadIdx.x & 31;
  const int wave  = threadIdx.x >> 5;     // 0..7
  const int waveM = wave >> 2;            // 0..1
  const int waveN = wave & 3;             // 0..3
  const int side  = blockIdx.z;
  const int b0    = blockIdx.x * 64 + waveM * 32;
  const int j0    = blockIdx.y * 128 + waveN * 32;
  const int half  = lane >> 4;            // 0 | 1
  const int l15   = lane & 15;

  const __bf16* aRow = xbf + ((size_t)side * BATCH + b0 + l15) * HRR;
  const __bf16* bRow = wt + (size_t)(j0 + l15) * HRR;
  const int aO0 = half * 8;               // per-lane K sub-offsets (A layout)
  const int aO1 = aO0 + 16;
  const int bO  = half * 8;               // per-lane K offset (B layout)

  v8f c00 = {}, c01 = {}, c10 = {}, c11 = {};

  union AB { v16bf v; U4 q[2]; };

#pragma unroll 2
  for (int kk = 0; kk < HRR; kk += 32) {
    AB a0, a1, bb0, bb1;
    a0.q[0]  = *(const U4*)(aRow + kk + aO0);
    a0.q[1]  = *(const U4*)(aRow + kk + aO1);
    a1.q[0]  = *(const U4*)(aRow + 16 * HRR + kk + aO0);
    a1.q[1]  = *(const U4*)(aRow + 16 * HRR + kk + aO1);
    bb0.q[0] = *(const U4*)(bRow + kk + bO);
    bb0.q[1] = *(const U4*)(bRow + kk + bO + 16);
    bb1.q[0] = *(const U4*)(bRow + 16 * HRR + kk + bO);
    bb1.q[1] = *(const U4*)(bRow + 16 * HRR + kk + bO + 16);

    c00 = __builtin_amdgcn_wmma_f32_16x16x32_bf16(false, a0.v, false, bb0.v, (short)0, c00, false, false);
    c01 = __builtin_amdgcn_wmma_f32_16x16x32_bf16(false, a0.v, false, bb1.v, (short)0, c01, false, false);
    c10 = __builtin_amdgcn_wmma_f32_16x16x32_bf16(false, a1.v, false, bb0.v, (short)0, c10, false, false);
    c11 = __builtin_amdgcn_wmma_f32_16x16x32_bf16(false, a1.v, false, bb1.v, (short)0, c11, false, false);
  }

  // Epilogue: C layout -> out[b, 2048 + side*16384 + j]
  float* o = out + 2 * HRR + (size_t)side * (NDEC * HRR) + j0 + l15;
#pragma unroll
  for (int v = 0; v < 8; ++v) {
    const size_t rm = (size_t)(b0 + half * 8 + v) * OUTW;
    o[rm]                          = c00[v];
    o[rm + 16]                     = c01[v];
    o[rm + (size_t)16 * OUTW]      = c10[v];
    o[rm + (size_t)16 * OUTW + 16] = c11[v];
  }
}

// ---------------------------------------------------------------------------
// Host launch. Workspace layout (needs 64 MiB):
//   ws[0          .. 32MiB) : Xbf  bf16 [2][8192][1024]
//   ws[32MiB      .. 64MiB) : Wt   bf16 [16384][1024]
// Both GEMM operands (64 MB total) fit in the 192 MB L2, so steady-state HBM
// traffic is essentially the 1.14 GB of output writes (~49 us floor @ 23.3 TB/s);
// the bf16 WMMA pipeline is what bounds the 5.5e11 FLops of matrix work.
// ---------------------------------------------------------------------------
extern "C" void kernel_launch(void* const* d_in, const int* in_sizes, int n_in,
                              void* d_out, int out_size, void* d_ws, size_t ws_size,
                              hipStream_t stream) {
  const float* prob = (const float*)d_in[0];
  const float* lem  = (const float*)d_in[1];
  const float* dec  = (const float*)d_in[2];
  float* out = (float*)d_out;

  __bf16* xbf = (__bf16*)d_ws;
  __bf16* wt  = (__bf16*)((char*)d_ws + 2 * SIDE_ELEMS * sizeof(__bf16)); // +32 MiB

  // 1) f32 passthrough + bf16 downconvert of X (both sides)
  xcvt_kernel<<<dim3((unsigned)(2 * SIDE_ELEMS / 4 / 256)), dim3(256), 0, stream>>>(
      prob, lem, xbf, out);

  // 2) materialize shifted-decoder matrix Wt (bf16, K-contiguous)
  wgen_kernel<<<dim3((unsigned)(W_ELEMS / 4 / 256)), dim3(256), 0, stream>>>(dec, wt);

  // 3) WMMA GEMM: grid (8192/64, 16384/128, 2 sides)
  hrr_gemm_kernel<<<dim3(BATCH / 64, (NDEC * HRR) / 128, 2), dim3(256), 0, stream>>>(
      xbf, wt, out);
}
